// DECODER_9371618640422
// MI455X (gfx1250) — compile-verified, matrix operand order checked by reading, not run on hardware
//
#include <hip/hip_runtime.h>
#include <hip/hip_bf16.h>

typedef __bf16 bf16;
typedef __attribute__((ext_vector_type(16))) __bf16 v16bf;
typedef __attribute__((ext_vector_type(8)))  __bf16 v8bf;
typedef __attribute__((ext_vector_type(8)))  float  v8f;

// ---------------- problem constants ----------------
#define NB    32
#define NTE   512
#define NTD   400
#define NENC  512
#define NDEC  1024
#define NATT  128
#define NFILT 128
#define NK    31
#define NPRE  256
#define NSPEC 80
#define NCTX  512

// ---------------- workspace layout (bytes) ----------------
static constexpr size_t OFF_BAR  = 0;                                     // 2 uints (padded 256)
static constexpr size_t OFF_PA   = 256;                                   // [32,512] f32
static constexpr size_t OFF_PC   = OFF_PA   + (size_t)NB*NTE*4;           // [32,512] f32
static constexpr size_t OFF_SC   = OFF_PC   + (size_t)NB*NTE*4;           // scores [32,512] f32
static constexpr size_t OFF_H    = OFF_SC   + (size_t)NB*NTE*4;           // h [32,1024] f32
static constexpr size_t OFF_HW2  = OFF_H    + (size_t)NB*NDEC*4;          // hW2 [32,128] f32
static constexpr size_t OFF_G1   = OFF_HW2  + (size_t)NB*NATT*4;          // [32,3072] f32
static constexpr size_t OFF_GI2  = OFF_G1   + (size_t)NB*3072*4;          // [32,3072] f32
static constexpr size_t OFF_GH2  = OFF_GI2  + (size_t)NB*3072*4;          // [32,3072] f32
static constexpr size_t OFF_EPRJ = OFF_GH2  + (size_t)NB*3072*4;          // enc_proj [32,512,128] f32
static constexpr size_t OFF_HBF  = OFF_EPRJ + (size_t)NB*NTE*NATT*4;      // h bf16 [32,1024]
static constexpr size_t OFF_X1   = OFF_HBF  + (size_t)NB*NDEC*2;          // xcat1 bf16 [32,768]
static constexpr size_t OFF_X2   = OFF_X1   + (size_t)NB*768*2;           // xcat2 bf16 [32,1536]
static constexpr size_t OFF_OC   = OFF_X2   + (size_t)NB*1536*2;          // outcat bf16 [32,1536]
static constexpr size_t OFF_EBF  = OFF_OC   + (size_t)NB*1536*2;          // encoder bf16 [32,512,512]
static constexpr size_t OFF_W1T  = OFF_EBF  + (size_t)NB*NTE*NENC*2;      // [128,512] bf16 (N,K)
static constexpr size_t OFF_W2T  = OFF_W1T  + (size_t)128*512*2;          // [128,1024]
static constexpr size_t OFF_LFT  = OFF_W2T  + (size_t)128*1024*2;         // loc_fc^T [128,128]
static constexpr size_t OFF_SWT  = OFF_LFT  + (size_t)128*128*2;          // spec_w^T [80,1536]
static constexpr size_t OFF_WIH1 = OFF_SWT  + (size_t)80*1536*2;          // [3072,768]
static constexpr size_t OFF_WIH2 = OFF_WIH1 + (size_t)3072*768*2;         // [3072,1536]
static constexpr size_t OFF_WHH2 = OFF_WIH2 + (size_t)3072*1536*2;        // [3072,1024]

// ---------------- WMMA fragment helpers ----------------
// A fragment (16x32 bf16), row-major source, lda in elements.
// lane<16: M=lane, K j<8 -> k0+j, j>=8 -> k0+j+8 ; lane>=16: M=lane-16, K offset +8.
__device__ __forceinline__ v16bf load_frag(const bf16* A, int lda, int k0) {
  int lane = threadIdx.x & 31;
  const bf16* p = A + (size_t)(lane & 15) * lda + k0 + ((lane >> 4) << 3);
  union { v16bf v; v8bf h[2]; } f;
  f.h[0] = *(const v8bf*)p;
  f.h[1] = *(const v8bf*)(p + 16);
  return f.v;
}

// C/D (16x16 f32): lane<16: n=lane, rows r ; lane>=16: n=lane-16, rows 8+r
__device__ __forceinline__ void store_tile(float* D, int ldd, int mbase, int nbase,
                                           v8f c, const float* bias) {
  int lane = threadIdx.x & 31;
  int n  = nbase + (lane & 15);
  int m0 = mbase + ((lane >> 4) << 3);
  float bv = bias ? bias[n] : 0.0f;
#pragma unroll
  for (int r = 0; r < 8; ++r) D[(size_t)(m0 + r) * ldd + n] = c[r] + bv;
}

// D[0:32, nbase:nbase+16] = A[32,K](bf16) @ W[N,K](bf16)^T + bias
__device__ __forceinline__ void gemm_m32(const bf16* A, int lda, const bf16* W, int ldk,
                                         int nbase, int K, const float* bias,
                                         float* D, int ldd) {
  v8f c0 = {}; v8f c1 = {};
  const bf16* Wn = W + (size_t)nbase * ldk;
  for (int k0 = 0; k0 < K; k0 += 32) {
    v16bf bb = load_frag(Wn, ldk, k0);
    v16bf a0 = load_frag(A, lda, k0);
    v16bf a1 = load_frag(A + (size_t)16 * lda, lda, k0);
    c0 = __builtin_amdgcn_wmma_f32_16x16x32_bf16(false, a0, false, bb, (short)0, c0, false, false);
    c1 = __builtin_amdgcn_wmma_f32_16x16x32_bf16(false, a1, false, bb, (short)0, c1, false, false);
  }
  store_tile(D, ldd, 0,  nbase, c0, bias);
  store_tile(D, ldd, 16, nbase, c1, bias);
}

// ---------------- misc device helpers ----------------
__device__ __forceinline__ float sigm(float x) { return 1.0f / (1.0f + __expf(-x)); }

__device__ __forceinline__ float rng01(unsigned a, unsigned b, unsigned c) {
  unsigned x = a * 0x9E3779B9u ^ b * 0x85EBCA6Bu ^ c * 0xC2B2AE35u;
  x ^= x >> 16; x *= 0x7FEB352Du; x ^= x >> 15; x *= 0x846CA68Bu; x ^= x >> 16;
  return (float)(x >> 8) * (1.0f / 16777216.0f);
}
__device__ __forceinline__ float dropf(float v, float p, unsigned t, unsigned idx, unsigned salt) {
  return (rng01(t, idx, salt) >= p) ? v / (1.0f - p) : 0.0f;
}

__device__ __forceinline__ void gbar(unsigned* bar) {
  __syncthreads();
  if (threadIdx.x == 0) {
    __threadfence();
    volatile unsigned* gen = bar + 1;
    unsigned target = *gen;
    if (atomicAdd(bar, 1u) == gridDim.x - 1) {
      *bar = 0;
      __threadfence();
      atomicAdd((unsigned*)(bar + 1), 1u);
    } else {
      while (*gen == target) { __builtin_amdgcn_s_sleep(2); }
    }
    __threadfence();
  }
  __syncthreads();
}

// ---------------- init: bf16 weight conversion / transposes / hidden0 ----------------
__global__ void k_prep(const float* __restrict__ encoder_op, const float* __restrict__ state,
                       const float* __restrict__ W1, const float* __restrict__ W2,
                       const float* __restrict__ loc_fc, const float* __restrict__ w_ih1,
                       const float* __restrict__ w_ih2, const float* __restrict__ w_hh2,
                       const float* __restrict__ spec_w, const float* __restrict__ bridge_w,
                       char* __restrict__ ws) {
  bf16* w1t  = (bf16*)(ws + OFF_W1T);
  bf16* w2t  = (bf16*)(ws + OFF_W2T);
  bf16* lft  = (bf16*)(ws + OFF_LFT);
  bf16* swt  = (bf16*)(ws + OFF_SWT);
  bf16* wi1  = (bf16*)(ws + OFF_WIH1);
  bf16* wi2  = (bf16*)(ws + OFF_WIH2);
  bf16* wh2  = (bf16*)(ws + OFF_WHH2);
  bf16* ebf  = (bf16*)(ws + OFF_EBF);
  float* hbuf = (float*)(ws + OFF_H);
  bf16*  hbf  = (bf16*)(ws + OFF_HBF);

  int g = blockIdx.x * blockDim.x + threadIdx.x;
  int st = gridDim.x * blockDim.x;

  for (int i = g; i < 128 * 512;  i += st) { int n = i / 512,  k = i % 512;  w1t[i] = (bf16)W1[k * 128 + n]; }
  for (int i = g; i < 128 * 1024; i += st) { int n = i / 1024, k = i % 1024; w2t[i] = (bf16)W2[k * 128 + n]; }
  for (int i = g; i < 128 * 128;  i += st) { int n = i >> 7,   k = i & 127;  lft[i] = (bf16)loc_fc[k * 128 + n]; }
  for (int i = g; i < 80 * 1536;  i += st) { int n = i / 1536, k = i % 1536; swt[i] = (bf16)spec_w[k * 80 + n]; }
  for (int i = g; i < 3072 * 768;  i += st) wi1[i] = (bf16)w_ih1[i];
  for (int i = g; i < 3072 * 1536; i += st) wi2[i] = (bf16)w_ih2[i];
  for (int i = g; i < 3072 * 1024; i += st) wh2[i] = (bf16)w_hh2[i];
  for (size_t i = g; i < (size_t)NB * NTE * NENC; i += st) ebf[i] = (bf16)encoder_op[i];

  // hidden0 = concat(state[0],state[1]) @ bridge_w
  for (int i = g; i < NB * NDEC; i += st) {
    int b = i >> 10, n = i & 1023;
    float acc = 0.0f;
    for (int k = 0; k < 256; ++k) acc += state[b * 256 + k] * bridge_w[k * 1024 + n];
    for (int k = 0; k < 256; ++k) acc += state[NB * 256 + b * 256 + k] * bridge_w[(256 + k) * 1024 + n];
    hbuf[i] = acc;
    hbf[i]  = (bf16)acc;
  }
}

// ---------------- enc_proj = encoder_op @ W1  (one-shot WMMA) ----------------
__global__ __launch_bounds__(256) void k_encproj(char* __restrict__ ws) {
  const bf16* ebf = (const bf16*)(ws + OFF_EBF);
  const bf16* w1t = (const bf16*)(ws + OFF_W1T);
  float* eprj = (float*)(ws + OFF_EPRJ);
  int mt = blockIdx.x;            // 0..1023  (B*TE/16 row tiles)
  int nt = threadIdx.x >> 5;      // 0..7     (ATT/16 col tiles, one per wave)
  const bf16* A = ebf + (size_t)mt * 16 * NENC;
  v8f c = {};
  for (int k0 = 0; k0 < NENC; k0 += 32) {
    v16bf bb = load_frag(w1t + (size_t)nt * 16 * NENC, NENC, k0);
    v16bf a  = load_frag(A, NENC, k0);
    c = __builtin_amdgcn_wmma_f32_16x16x32_bf16(false, a, false, bb, (short)0, c, false, false);
  }
  store_tile(eprj + (size_t)mt * 16 * NATT, NATT, 0, nt * 16, c, nullptr);
}

// ---------------- persistent decoder ----------------
__global__ __launch_bounds__(256, 1) void k_decoder(
    const int* __restrict__ maskp, const float* __restrict__ decoder_ip,
    const float* __restrict__ encoder_op, const float* __restrict__ conv_w,
    const float* __restrict__ Vv, const float* __restrict__ p1w, const float* __restrict__ p2w,
    const float* __restrict__ b_ih1, const float* __restrict__ b_hh1,
    const float* __restrict__ b_ih2, const float* __restrict__ b_hh2,
    const float* __restrict__ stop_w, char* __restrict__ ws, float* __restrict__ out) {

  unsigned* bar = (unsigned*)(ws + OFF_BAR);
  float* pa    = (float*)(ws + OFF_PA);
  float* pc    = (float*)(ws + OFF_PC);
  float* scr   = (float*)(ws + OFF_SC);
  float* hbuf  = (float*)(ws + OFF_H);
  float* hW2v  = (float*)(ws + OFF_HW2);
  float* g1    = (float*)(ws + OFF_G1);
  float* gi2   = (float*)(ws + OFF_GI2);
  float* gh2   = (float*)(ws + OFF_GH2);
  float* eprj  = (float*)(ws + OFF_EPRJ);
  bf16*  hbf   = (bf16*)(ws + OFF_HBF);
  bf16*  x1bf  = (bf16*)(ws + OFF_X1);
  bf16*  x2bf  = (bf16*)(ws + OFF_X2);
  bf16*  ocbf  = (bf16*)(ws + OFF_OC);
  const bf16* w2t = (const bf16*)(ws + OFF_W2T);
  const bf16* lft = (const bf16*)(ws + OFF_LFT);
  const bf16* swt = (const bf16*)(ws + OFF_SWT);
  const bf16* wi1 = (const bf16*)(ws + OFF_WIH1);
  const bf16* wi2 = (const bf16*)(ws + OFF_WIH2);
  const bf16* wh2 = (const bf16*)(ws + OFF_WHH2);

  const int tid  = threadIdx.x;
  const int lane = tid & 31;
  const int wid  = tid >> 5;
  const int bidx = blockIdx.x;              // grid is exactly 32 blocks
  const int gw   = bidx * 8 + wid;          // global wave id 0..255
  const int gtid = bidx * 256 + tid;        // global thread id 0..8191

  static __shared__ union {
    bf16 tiles[8][2048];                    // per-wave 16x128 conv tile
    struct { float red[256]; float sprob[512]; } s2;
    float y1[256];
  } sm;

#pragma unroll 1
  for (int t = 0; t < NTD; ++t) {
    // ---- S0: prenet (b = bidx), spec/stop of previous step, hW2 = h @ W2 ----
    {
      int b = bidx;
      float y1v = 0.0f;
      if (t > 0) {
        const float* curr = decoder_ip + ((size_t)b * NTD + (t - 1)) * NSPEC;
        for (int k = 0; k < NSPEC; ++k) y1v += curr[k] * p1w[k * NPRE + tid];
      }
      y1v = fmaxf(y1v, 0.0f);
      y1v = dropf(y1v, 0.5f, (unsigned)t, (unsigned)(b * 256 + tid), 0x1111u);
      sm.y1[tid] = y1v;
      __syncthreads();
      float y2v = 0.0f;
      for (int k = 0; k < NPRE; ++k) y2v += sm.y1[k] * p2w[k * NPRE + tid];
      y2v = fmaxf(y2v, 0.0f);
      y2v = dropf(y2v, 0.5f, (unsigned)t, (unsigned)(b * 256 + tid), 0x2222u);
      x1bf[b * 768 + tid] = (bf16)y2v;
      __syncthreads();
    }
    if (t > 0) {
      int tp = t - 1;
      if (gw < 5) {                          // spec = outcat @ spec_w  (N=80 -> 5 tiles)
        gemm_m32(ocbf, 1536, swt, 1536, gw * 16, 1536, nullptr,
                 out + (size_t)tp * NSPEC, NTD * NSPEC);
      } else if (gw == 5) {                  // stop scalar
        int b = lane;
        float acc = 0.0f;
        for (int k = 0; k < 1536; ++k) acc += (float)ocbf[b * 1536 + k] * stop_w[k];
        out[(size_t)NB * NTD * NSPEC + b * NTD + tp] = sigm(acc);
      }
    }
    if (gw < 8) {                            // hW2 [32,128]
      gemm_m32(hbf, NDEC, w2t, NDEC, gw * 16, NDEC, nullptr, hW2v, NATT);
    }
    gbar(bar);

    // ---- P1: location conv + loc_fc WMMA + tanh()*V scores ----
    for (int item = gw; item < NB * 32; item += 256) {   // exactly 4 iters per wave
      int b = item >> 5;
      int te0 = (item & 31) << 4;
      bf16* tile = sm.tiles[wid];
      for (int idx = lane; idx < 2048; idx += 32) {
        int ter = idx >> 7, f = idx & 127;
        const float* w = conv_w + f * (2 * NK);
        int tb = te0 + ter - (NK - 1) / 2;
        float acc = 0.0f;
#pragma unroll 1
        for (int k = 0; k < NK; ++k) {
          int tt = tb + k;
          if (tt >= 0 && tt < NTE)
            acc += w[k] * pa[b * NTE + tt] + w[NK + k] * pc[b * NTE + tt];
        }
        tile[idx] = (bf16)acc;
      }
      __syncthreads();
      float accs[8];
#pragma unroll
      for (int r = 0; r < 8; ++r) accs[r] = 0.0f;
      int n_lane = lane & 15;
      int mhalf = (lane >> 4) << 3;
      for (int nt = 0; nt < 8; ++nt) {
        v8f c = {};
        for (int k0 = 0; k0 < NFILT; k0 += 32) {
          v16bf bb = load_frag(lft + (size_t)nt * 16 * NFILT, NFILT, k0);
          v16bf a  = load_frag(tile, NFILT, k0);
          c = __builtin_amdgcn_wmma_f32_16x16x32_bf16(false, a, false, bb, (short)0, c, false, false);
        }
        int n = nt * 16 + n_lane;
        float vv = Vv[n];
        float hw = hW2v[b * NATT + n];
#pragma unroll
        for (int r = 0; r < 8; ++r) {
          int te = te0 + mhalf + r;
          float s = tanhf(eprj[((size_t)(b * NTE + te)) * NATT + n] + hw + c[r]);
          accs[r] += s * vv;
        }
      }
#pragma unroll
      for (int r = 0; r < 8; ++r) {
        float s = accs[r];
        s += __shfl_xor(s, 1, 32);
        s += __shfl_xor(s, 2, 32);
        s += __shfl_xor(s, 4, 32);
        s += __shfl_xor(s, 8, 32);
        if (n_lane == 0) {
          int te = te0 + mhalf + r;
          scr[b * NTE + te] = (maskp[b * NTE + te] != 0) ? s : -1e10f;
        }
      }
      __syncthreads();
    }
    gbar(bar);

    // ---- P2: masked softmax + ctx (one block per batch row) ----
    {
      int b = bidx;
      float s0 = scr[b * NTE + tid];
      float s1 = scr[b * NTE + 256 + tid];
      sm.s2.red[tid] = fmaxf(s0, s1);
      __syncthreads();
      for (int off = 128; off > 0; off >>= 1) {
        if (tid < off) sm.s2.red[tid] = fmaxf(sm.s2.red[tid], sm.s2.red[tid + off]);
        __syncthreads();
      }
      float mx = sm.s2.red[0];
      __syncthreads();
      float e0 = __expf(s0 - mx), e1 = __expf(s1 - mx);
      sm.s2.red[tid] = e0 + e1;
      __syncthreads();
      for (int off = 128; off > 0; off >>= 1) {
        if (tid < off) sm.s2.red[tid] += sm.s2.red[tid + off];
        __syncthreads();
      }
      float inv = 1.0f / sm.s2.red[0];
      __syncthreads();
      float q0 = e0 * inv, q1 = e1 * inv;
      sm.s2.sprob[tid] = q0; sm.s2.sprob[tid + 256] = q1;
      pa[b * NTE + tid] = q0;       pa[b * NTE + 256 + tid] = q1;
      pc[b * NTE + tid] += q0;      pc[b * NTE + 256 + tid] += q1;
      __syncthreads();
      for (int e = tid; e < NENC; e += 256) {
        float acc = 0.0f;
#pragma unroll 4
        for (int tt = 0; tt < NTE; ++tt)
          acc += sm.s2.sprob[tt] * encoder_op[((size_t)b * NTE + tt) * NENC + e];
        bf16 cb = (bf16)acc;
        x1bf[b * 768 + 256 + e]   = cb;
        x2bf[b * 1536 + 1024 + e] = cb;
        ocbf[b * 1536 + 1024 + e] = cb;
      }
    }
    gbar(bar);

    // ---- P3: GRU1 input gemm: g1 = xcat1 @ w_ih1^T + b_ih1 ----
    if (gw < 192) gemm_m32(x1bf, 768, wi1, 768, gw * 16, 768, b_ih1, g1, 3072);
    gbar(bar);

    // ---- P3b: GRU1 elementwise (h0 = 0 -> gh = b_hh1) ----
    for (int idx = gtid; idx < NB * NDEC; idx += 8192) {
      int b = idx >> 10, n = idx & 1023;
      const float* gi = g1 + b * 3072;
      float r  = sigm(gi[n] + b_hh1[n]);
      float z  = sigm(gi[1024 + n] + b_hh1[1024 + n]);
      float nn = tanhf(gi[2048 + n] + r * b_hh1[2048 + n]);
      float h1 = (1.0f - z) * nn;
      h1 = dropf(h1, 0.1f, (unsigned)t, (unsigned)idx, 0x3333u);
      x2bf[b * 1536 + n] = (bf16)h1;
    }
    gbar(bar);

    // ---- P4: GRU2 gemms: gi2 = xcat2 @ w_ih2^T + b_ih2 ; gh2 = h @ w_hh2^T + b_hh2 ----
    for (int tile = gw; tile < 384; tile += 256) {
      if (tile < 192) gemm_m32(x2bf, 1536, wi2, 1536, tile * 16, 1536, b_ih2, gi2, 3072);
      else            gemm_m32(hbf,  NDEC, wh2, NDEC, (tile - 192) * 16, NDEC, b_hh2, gh2, 3072);
    }
    gbar(bar);

    // ---- P4b: GRU2 elementwise -> new h, outcat ----
    for (int idx = gtid; idx < NB * NDEC; idx += 8192) {
      int b = idx >> 10, n = idx & 1023;
      const float* gi = gi2 + b * 3072;
      const float* gh = gh2 + b * 3072;
      float r  = sigm(gi[n] + gh[n]);
      float z  = sigm(gi[1024 + n] + gh[1024 + n]);
      float nn = tanhf(gi[2048 + n] + r * gh[2048 + n]);
      float hp = hbuf[idx];
      float hn = (1.0f - z) * nn + z * hp;
      hbuf[idx] = hn;
      hbf[idx]  = (bf16)hn;
      float hd = dropf(hn, 0.1f, (unsigned)t, (unsigned)idx, 0x4444u);
      ocbf[b * 1536 + n] = (bf16)hd;
    }
    gbar(bar);
  }

  // ---- final step's spec/stop ----
  {
    int tp = NTD - 1;
    if (gw < 5) {
      gemm_m32(ocbf, 1536, swt, 1536, gw * 16, 1536, nullptr,
               out + (size_t)tp * NSPEC, NTD * NSPEC);
    } else if (gw == 5) {
      int b = lane;
      float acc = 0.0f;
      for (int k = 0; k < 1536; ++k) acc += (float)ocbf[b * 1536 + k] * stop_w[k];
      out[(size_t)NB * NTD * NSPEC + b * NTD + tp] = sigm(acc);
    }
  }
}

extern "C" void kernel_launch(void* const* d_in, const int* in_sizes, int n_in,
                              void* d_out, int out_size, void* d_ws, size_t ws_size,
                              hipStream_t stream) {
  (void)in_sizes; (void)n_in; (void)out_size; (void)ws_size;
  const float* encoder_op = (const float*)d_in[0];
  const float* state      = (const float*)d_in[1];
  const float* decoder_ip = (const float*)d_in[2];
  const int*   mask       = (const int*)  d_in[3];
  const float* W1         = (const float*)d_in[4];
  const float* W2         = (const float*)d_in[5];
  const float* V          = (const float*)d_in[6];
  const float* conv_w     = (const float*)d_in[7];
  const float* loc_fc     = (const float*)d_in[8];
  const float* p1         = (const float*)d_in[9];
  const float* p2         = (const float*)d_in[10];
  const float* w_ih1      = (const float*)d_in[11];
  /* w_ih1's hidden gemm vanishes (h0==0): d_in[12] (w_hh1) unused */
  const float* b_ih1      = (const float*)d_in[13];
  const float* b_hh1      = (const float*)d_in[14];
  const float* w_ih2      = (const float*)d_in[15];
  const float* w_hh2      = (const float*)d_in[16];
  const float* b_ih2      = (const float*)d_in[17];
  const float* b_hh2      = (const float*)d_in[18];
  const float* spec_w     = (const float*)d_in[19];
  const float* stop_w     = (const float*)d_in[20];
  const float* bridge_w   = (const float*)d_in[21];
  char* ws = (char*)d_ws;
  float* out = (float*)d_out;

  // zero barrier counters + pa + pc (contiguous prefix of ws)
  hipMemsetAsync(d_ws, 0, OFF_SC, stream);
  k_prep<<<2048, 256, 0, stream>>>(encoder_op, state, W1, W2, loc_fc, w_ih1,
                                   w_ih2, w_hh2, spec_w, bridge_w, ws);
  k_encproj<<<(NB * NTE) / 16, 256, 0, stream>>>(ws);
  k_decoder<<<32, 256, 0, stream>>>(mask, decoder_ip, encoder_op, conv_w, V, p1, p2,
                                    b_ih1, b_hh1, b_ih2, b_hh2, stop_w, ws, out);
}
